// GlobalLocalAttention_45200235823168
// MI455X (gfx1250) — matrix operand order repaired and checked by reading, not compiled, not run on hardware
//
#include <hip/hip_runtime.h>

// ---------------------------------------------------------------------------
// MI455X (gfx1250) Swin global-local attention block.
// NHWC bf16 activations internally; all GEMM-shaped work on
// v_wmma_f32_16x16x32_bf16; GEMM tile staging via the Tensor Data Mover
// (tensor_load_to_lds + s_wait_tensorcnt) with LDS row padding done by TDM.
// ---------------------------------------------------------------------------

typedef __bf16 bf16;
typedef __attribute__((ext_vector_type(16))) __bf16 v16bf;
typedef __attribute__((ext_vector_type(8)))  __bf16 v8bf;
typedef __attribute__((ext_vector_type(8)))  float  v8f;
typedef __attribute__((ext_vector_type(4)))  unsigned int v4u;
typedef __attribute__((ext_vector_type(8)))  int v8i;
typedef __attribute__((ext_vector_type(4)))  int v4i;

#define BN_INV 0.9999950000374997f     // 1/sqrt(1 + 1e-5)
#define ATT_SCALE 0.17677669529663687f // 32^-0.5

#if __has_builtin(__builtin_amdgcn_tensor_load_to_lds) && \
    __has_builtin(__builtin_amdgcn_s_wait_tensorcnt)
#define HAVE_TDM 1
#else
#define HAVE_TDM 0
#endif

union V16U { v16bf v; v8bf h[2]; };

static __device__ __forceinline__ v8f vzero8() {
  v8f z;
#pragma unroll
  for (int i = 0; i < 8; ++i) z[i] = 0.0f;
  return z;
}
static __device__ __forceinline__ v8bf bzero8() {
  v8bf z;
#pragma unroll
  for (int i = 0; i < 8; ++i) z[i] = (bf16)0.0f;
  return z;
}

static __device__ __forceinline__ v8f wmma_bf16(v16bf a, v16bf b, v8f c) {
  return __builtin_amdgcn_wmma_f32_16x16x32_bf16(
      false, a, false, b, (short)0, c, false, false);
}

// A fragment (16x32 MxK): lane m = lane&15, ko = (lane>>4)*8;
// elems 0..7 = K[ko..ko+7], elems 8..15 = K[ko+16..ko+23].
static __device__ __forceinline__ v16bf frag_a(const bf16* tile, int stride, int kbase) {
  int lane = threadIdx.x & 31;
  const bf16* p = tile + (lane & 15) * stride + kbase + ((lane >> 4) << 3);
  V16U u;
  u.h[0] = *(const v8bf*)p;
  u.h[1] = *(const v8bf*)(p + 16);
  return u.v;
}

// B fragment (32x16 KxN) from transposed tile bT[n][k]:
// lane n = lane&15, ko = kbase + (lane>>4)*8; elems 0..7 = K[ko..ko+7], elems 8..15 = K[ko+16..ko+23].
static __device__ __forceinline__ v16bf frag_b(const bf16* tileT, int stride, int kbase) {
  int lane = threadIdx.x & 31;
  const bf16* p = tileT + (lane & 15) * stride + kbase + ((lane >> 4) << 3);
  V16U u;
  u.h[0] = *(const v8bf*)p;
  u.h[1] = *(const v8bf*)(p + 16);
  return u.v;
}

#if HAVE_TDM
// Issue one 2D TDM tile load: tile_d1 rows of tile_d0 bf16 elements, global
// row stride = stride0 elements, LDS rows padded by hardware to 72 elements
// (pad_interval code 4 = 32 DWORDs = one 64-elem row, pad_amount code 3 =
// 4 DWORDs = 8 elems). Descriptors per CDNA5 ISA.
static __device__ __forceinline__ void tdm_load_2d(
    unsigned lds_byte, unsigned long long gaddr_byte,
    unsigned tensor_d0, unsigned tensor_d1,
    unsigned tile_d0, unsigned tile_d1, unsigned stride0) {
  v4u g0;
  g0[0] = 1u;                                            // count=1, load, no gather
  g0[1] = lds_byte;                                      // lds_addr
  g0[2] = (unsigned)gaddr_byte;                          // global_addr[31:0]
  g0[3] = (unsigned)((gaddr_byte >> 32) & 0x1FFFFFFull)  // global_addr[56:32]
          | (2u << 30);                                  // type = 2 ("image")
  v8i g1;
  g1[0] = (int)((1u << 16)      // data_size = 1 (2 bytes)
                | (1u << 20)    // pad_enable
                | (4u << 22)    // pad_interval: 32 DWORDs (one 64-elem row)
                | (3u << 25));  // pad_amount: 4 DWORDs (8 elems)
  g1[1] = (int)(tensor_d0 << 16);
  g1[2] = (int)((tensor_d0 >> 16) | (tensor_d1 << 16));
  g1[3] = (int)((tensor_d1 >> 16) | (tile_d0 << 16));
  g1[4] = (int)(tile_d1 & 0xFFFFu);                      // tile_dim1 (tile_dim2 = 0)
  g1[5] = (int)stride0;                                  // tensor_dim0_stride[31:0]
  g1[6] = 0;
  g1[7] = 0;
  v4i gz = {0, 0, 0, 0};
#if __clang_major__ >= 23
  v8i gz8 = {0, 0, 0, 0, 0, 0, 0, 0};
  __builtin_amdgcn_tensor_load_to_lds(g0, g1, gz, gz, gz8, 0);
#else
  __builtin_amdgcn_tensor_load_to_lds(g0, g1, gz, gz, 0);
#endif
}
#endif

// ---------------------------------------------------------------------------
// WMMA GEMM: out[m][n] = sum_k A[m][k] * XT[n][k]   (XT = NHWC activations)
// 256 threads = 8 waves (4x2), 128x64 tile, K-step 64. Tile staging by TDM.
// TAPS==9: implicit 3x3 conv (A tiles via TDM, shifted B rows staged manually
// with zero borders). EPI: 0 = f32 NCHW, 1 = bf16 NHWC, 2 = BN NHWC store,
// 3 = BN NHWC accumulate.
// ---------------------------------------------------------------------------
template <int EPI, int TAPS>
__global__ __launch_bounds__(256) void gemm_kernel(
    const bf16* __restrict__ A,    // [TAPS][M][512]
    const bf16* __restrict__ XT,   // [b][4096][512]
    float* __restrict__ outF, bf16* __restrict__ outH,
    const float* __restrict__ gamma, const float* __restrict__ beta, int M) {
  const int K = 512;
  __shared__ __align__(16) bf16 sA[128 * 72];   // A tile  [m][k], 72-elem rows
  __shared__ __align__(16) bf16 sBT[64 * 72];   // X tile  [n][k], 72-elem rows

  int tid = threadIdx.x;
  int n0 = blockIdx.x * 64;      // one full image row (W = 64)
  int m0 = blockIdx.y * 128;
  int b = blockIdx.z;
  int h = blockIdx.x;
  const bf16* Xb = XT + (size_t)b * 4096 * 512;

  int wave = tid >> 5, lane = tid & 31;
  int wm = wave >> 1, wn = wave & 1;

  v8f acc[2][2];
#pragma unroll
  for (int i = 0; i < 2; ++i)
#pragma unroll
    for (int j = 0; j < 2; ++j) acc[i][j] = vzero8();

  int am = tid & 127, akc = (tid >> 7) * 32;   // manual A staging coords
  int bn = tid & 63,  bkc = (tid >> 6) * 16;   // manual B staging coords

  for (int t = 0; t < TAPS; ++t) {
    int dh = (TAPS == 9) ? (t / 3 - 1) : 0;
    int dw = (TAPS == 9) ? (t % 3 - 1) : 0;
    int sh = h + dh, sw = bn + dw;
    bool valid = ((unsigned)sh < 64u) && ((unsigned)sw < 64u);
    const bf16* Abase = A + (size_t)t * M * K + (size_t)m0 * K;   // uniform
    const bf16* Arow = Abase + (size_t)am * K + akc;              // per-thread
    const bf16* Bbase = Xb + (size_t)(h * 64) * 512;              // uniform
    const bf16* Brow = Xb + (size_t)(sh * 64 + sw) * 512 + bkc;   // per-thread

    for (int k0 = 0; k0 < K; k0 += 64) {
#if HAVE_TDM
      if (tid == 0) {
        // A tile: 128 rows x 64 elems
        tdm_load_2d((unsigned)(size_t)&sA[0],
                    (unsigned long long)(size_t)(const void*)(Abase + k0),
                    512u, (unsigned)M, 64u, 128u, 512u);
        if (TAPS == 1)  // B tile: 64 pixel rows x 64 elems (always in-image)
          tdm_load_2d((unsigned)(size_t)&sBT[0],
                      (unsigned long long)(size_t)(const void*)(Bbase + k0),
                      512u, 4096u, 64u, 64u, 512u);
      }
      if (TAPS == 9) {  // shifted B rows: manual, zero outside image
        v8bf r0, r1;
        if (valid) {
          const v8bf* s = (const v8bf*)(Brow + k0);
          r0 = s[0]; r1 = s[1];
        } else {
          r0 = bzero8(); r1 = bzero8();
        }
        v8bf* d = (v8bf*)&sBT[bn * 72 + bkc];
        d[0] = r0; d[1] = r1;
      }
      if (tid == 0) __builtin_amdgcn_s_wait_tensorcnt((short)0);
#else
      {  // manual A staging: 4 x 16B
        const v8bf* s = (const v8bf*)(Arow + k0);
        v8bf r0 = s[0], r1 = s[1], r2 = s[2], r3 = s[3];
        v8bf* d = (v8bf*)&sA[am * 72 + akc];
        d[0] = r0; d[1] = r1; d[2] = r2; d[3] = r3;
        if (k0 + 64 < K) __builtin_prefetch(Arow + k0 + 64, 0, 0);
      }
      {  // manual B staging: 2 x 16B
        v8bf r0, r1;
        if (valid) {
          const v8bf* s = (const v8bf*)(Brow + k0);
          r0 = s[0]; r1 = s[1];
          if (k0 + 64 < K) __builtin_prefetch(Brow + k0 + 64, 0, 0);
        } else {
          r0 = bzero8(); r1 = bzero8();
        }
        v8bf* d = (v8bf*)&sBT[bn * 72 + bkc];
        d[0] = r0; d[1] = r1;
      }
#endif
      __syncthreads();
#pragma unroll
      for (int kb = 0; kb < 64; kb += 32) {
        v16bf af0 = frag_a(&sA[(wm * 32) * 72], 72, kb);
        v16bf af1 = frag_a(&sA[(wm * 32 + 16) * 72], 72, kb);
        v16bf bg0 = frag_b(&sBT[(wn * 32) * 72], 72, kb);
        v16bf bg1 = frag_b(&sBT[(wn * 32 + 16) * 72], 72, kb);
        acc[0][0] = wmma_bf16(af0, bg0, acc[0][0]);
        acc[0][1] = wmma_bf16(af0, bg1, acc[0][1]);
        acc[1][0] = wmma_bf16(af1, bg0, acc[1][0]);
        acc[1][1] = wmma_bf16(af1, bg1, acc[1][1]);
      }
      __syncthreads();
    }
  }

  // epilogue: C layout -> lane holds fixed n, 8 consecutive m per half-wave
  int nl = lane & 15, rhi = (lane >> 4) << 3;
#pragma unroll
  for (int i = 0; i < 2; ++i)
#pragma unroll
    for (int j = 0; j < 2; ++j) {
      int mbase = m0 + wm * 32 + i * 16 + rhi;
      int nb = n0 + wn * 32 + j * 16 + nl;
      if (EPI == 0) {                       // NCHW f32 (final output)
#pragma unroll
        for (int r = 0; r < 8; ++r)
          outF[((size_t)b * M + mbase + r) * 4096 + nb] = acc[i][j][r];
      } else if (EPI == 1) {                // NHWC bf16 (qkv), contiguous in m
        v8bf pk;
#pragma unroll
        for (int r = 0; r < 8; ++r) pk[r] = (bf16)acc[i][j][r];
        *(v8bf*)&outH[((size_t)(b * 4096 + nb)) * M + mbase] = pk;
      } else {                              // BN epilogues, NHWC f32
        v8f gg = *(const v8f*)&gamma[mbase];
        v8f bb = *(const v8f*)&beta[mbase];
        v8f res;
#pragma unroll
        for (int r = 0; r < 8; ++r)
          res[r] = gg[r] * (acc[i][j][r] * BN_INV) + bb[r];
        v8f* p = (v8f*)&outF[((size_t)(b * 4096 + nb)) * M + mbase];
        if (EPI == 2) *p = res;
        else          *p = *p + res;
      }
    }
}

// ---------------------------------------------------------------------------
// Window attention: one block per (window, head); NHWC qkv so Q/K rows are
// 64B-contiguous. 4 waves own 16 query rows each: S = Q@K^T, in-register
// softmax (half-wave shuffles), P@V, coalesced NHWC output stores.
// ---------------------------------------------------------------------------
__global__ __launch_bounds__(128) void attn_kernel(
    const bf16* __restrict__ qkvT, const float* __restrict__ rel_table,
    float* __restrict__ attT) {
  __shared__ __align__(16) float bias[64 * 64];
  __shared__ __align__(16) bf16 qs[64 * 40];    // Q[l][d]
  __shared__ __align__(16) bf16 ks[64 * 40];    // K[j][d]
  __shared__ __align__(16) bf16 vts[32 * 72];   // V^T[d][j]
  __shared__ __align__(16) bf16 ps[64 * 72];    // P[l][j]

  int tid = threadIdx.x;
  int widx = blockIdx.x, head = blockIdx.y;
  int b = widx >> 6;
  int hh = (widx >> 3) & 7, ww = widx & 7;

  {  // stage Q and K: 128 tasks, one 64B row each
    int mat = tid >> 6, l = tid & 63;
    int px = (hh * 8 + (l >> 3)) * 64 + ww * 8 + (l & 7);
    const bf16* src = qkvT + ((size_t)(b * 4096 + px)) * 1536 + mat * 512 + head * 32;
    bf16* dst = (mat ? ks : qs) + l * 40;
    const v8bf* s8 = (const v8bf*)src;
    v8bf* d8 = (v8bf*)dst;
    d8[0] = s8[0]; d8[1] = s8[1]; d8[2] = s8[2]; d8[3] = s8[3];
  }
  // stage V transposed (coalesced 64B reads, scalar LDS scatter)
  for (int e = tid; e < 2048; e += 128) {
    int l = e >> 5, d = e & 31;
    int px = (hh * 8 + (l >> 3)) * 64 + ww * 8 + (l & 7);
    vts[d * 72 + l] = qkvT[((size_t)(b * 4096 + px)) * 1536 + 1024 + head * 32 + d];
  }
  // stage Swin relative-position bias
  for (int e = tid; e < 4096; e += 128) {
    int i = e >> 6, j = e & 63;
    int ry = (i >> 3) - (j >> 3) + 7;
    int rx = (i & 7) - (j & 7) + 7;
    bias[e] = rel_table[(ry * 15 + rx) * 16 + head];
  }
  __syncthreads();

  int wave = tid >> 5, lane = tid & 31;
  int nl = lane & 15, rhi = (lane >> 4) << 3;

  // S = Q @ K^T (HD = 32 -> single WMMA K-step)
  v16bf aq = frag_a(&qs[(wave * 16) * 40], 40, 0);
  v8f s[4];
#pragma unroll
  for (int nc = 0; nc < 4; ++nc)
    s[nc] = wmma_bf16(aq, frag_b(&ks[(nc * 16) * 40], 40, 0), vzero8());

  // softmax per row (each row lives in one 16-lane half across 4 N-tiles)
#pragma unroll
  for (int r = 0; r < 8; ++r) {
    int row = wave * 16 + r + rhi;
    float v0 = s[0][r] * ATT_SCALE + bias[row * 64 + 0 + nl];
    float v1 = s[1][r] * ATT_SCALE + bias[row * 64 + 16 + nl];
    float v2 = s[2][r] * ATT_SCALE + bias[row * 64 + 32 + nl];
    float v3 = s[3][r] * ATT_SCALE + bias[row * 64 + 48 + nl];
    float mx = fmaxf(fmaxf(v0, v1), fmaxf(v2, v3));
#pragma unroll
    for (int o = 1; o < 16; o <<= 1) mx = fmaxf(mx, __shfl_xor(mx, o, 32));
    float e0 = __expf(v0 - mx), e1 = __expf(v1 - mx);
    float e2 = __expf(v2 - mx), e3 = __expf(v3 - mx);
    float sum = e0 + e1 + e2 + e3;
#pragma unroll
    for (int o = 1; o < 16; o <<= 1) sum += __shfl_xor(sum, o, 32);
    float inv = 1.0f / sum;
    ps[row * 72 + 0 + nl]  = (bf16)(e0 * inv);
    ps[row * 72 + 16 + nl] = (bf16)(e1 * inv);
    ps[row * 72 + 32 + nl] = (bf16)(e2 * inv);
    ps[row * 72 + 48 + nl] = (bf16)(e3 * inv);
  }
  __syncthreads();

  // O = P @ V : 16 rows x HD=32, K = 64 (2 steps)
  v8f o[2] = {vzero8(), vzero8()};
#pragma unroll
  for (int kstep = 0; kstep < 2; ++kstep) {
    v16bf ap = frag_a(&ps[(wave * 16) * 72], 72, kstep * 32);
#pragma unroll
    for (int nc = 0; nc < 2; ++nc)
      o[nc] = wmma_bf16(ap, frag_b(&vts[(nc * 16) * 72], 72, kstep * 32), o[nc]);
  }

  // NHWC store: lanes cover 16 consecutive channels -> coalesced
#pragma unroll
  for (int nc = 0; nc < 2; ++nc)
#pragma unroll
    for (int r = 0; r < 8; ++r) {
      int d = nc * 16 + nl;
      int l = wave * 16 + r + rhi;
      int px = (hh * 8 + (l >> 3)) * 64 + ww * 8 + (l & 7);
      attT[((size_t)(b * 4096 + px)) * 512 + head * 32 + d] = o[nc][r];
    }
}

// ---------------------------------------------------------------------------
// NHWC pooling / depthwise (fp32 VALU, channel-fastest = coalesced)
// ---------------------------------------------------------------------------
static __device__ __forceinline__ float srcpixT(const float* base, int i, int j, int c) {
  int si = (i == 64) ? 62 : i;
  int sj = (j == 64) ? 62 : j;
  if ((unsigned)si > 63u || (unsigned)sj > 63u) return 0.0f;
  return base[((size_t)(si * 64 + sj)) * 512 + c];
}

__global__ __launch_bounds__(256) void pool_kernel(
    const float* __restrict__ attT, const float* __restrict__ localT,
    float* __restrict__ mixT) {
  int i = blockIdx.x * 256 + threadIdx.x;   // over 8*4096*512, c fastest
  int c = i & 511;
  int bpx = i >> 9;
  int px = bpx & 4095, b = bpx >> 12;
  int h = px >> 6, w = px & 63;
  const float* base = attT + ((size_t)b * 4096) * 512;
  float s = 0.0f;
#pragma unroll
  for (int t = 0; t < 8; ++t) s += srcpixT(base, h - 3 + t, w, c);
#pragma unroll
  for (int t = 0; t < 8; ++t) s += srcpixT(base, h, w - 3 + t, c);
  mixT[i] = s * 0.125f + localT[i];
}

__global__ __launch_bounds__(256) void dwconv_kernel(
    const float* __restrict__ mixT, const float* __restrict__ wdwT,
    const float* __restrict__ g3, const float* __restrict__ b3,
    bf16* __restrict__ outT) {
  int i = blockIdx.x * 256 + threadIdx.x;   // over 8*4096*128 (4 ch/thread)
  int cq = i & 127;
  int bpx = i >> 7;
  int px = bpx & 4095, b = bpx >> 12;
  int h = px >> 6, w = px & 63;
  int c0 = cq * 4;
  const float* mb = mixT + ((size_t)b * 4096) * 512;

  int sis[8], sjs[8];
#pragma unroll
  for (int t = 0; t < 8; ++t) {
    int ii = h - 3 + t;
    int si = (ii == 64) ? 62 : ii;
    sis[t] = ((unsigned)si > 63u) ? -1 : si;
    int jj = w - 3 + t;
    int sj = (jj == 64) ? 62 : jj;
    sjs[t] = ((unsigned)sj > 63u) ? -1 : sj;
  }
  float a0 = 0.f, a1 = 0.f, a2 = 0.f, a3 = 0.f;
#pragma unroll
  for (int th = 0; th < 8; ++th) {
    if (sis[th] < 0) continue;
#pragma unroll
    for (int tw = 0; tw < 8; ++tw) {
      if (sjs[tw] < 0) continue;
      float4 pv = *(const float4*)&mb[((size_t)(sis[th] * 64 + sjs[tw])) * 512 + c0];
      float4 wv = *(const float4*)&wdwT[(th * 8 + tw) * 512 + c0];
      a0 += pv.x * wv.x; a1 += pv.y * wv.y; a2 += pv.z * wv.z; a3 += pv.w * wv.w;
    }
  }
  float4 gv = *(const float4*)&g3[c0];
  float4 bv = *(const float4*)&b3[c0];
  bf16* out = outT + (size_t)bpx * 512 + c0;
  out[0] = (bf16)(gv.x * (a0 * BN_INV) + bv.x);
  out[1] = (bf16)(gv.y * (a1 * BN_INV) + bv.y);
  out[2] = (bf16)(gv.z * (a2 * BN_INV) + bv.z);
  out[3] = (bf16)(gv.w * (a3 * BN_INV) + bv.w);
}

// ---------------------------------------------------------------------------
// Conversion / repack kernels
// ---------------------------------------------------------------------------
// NCHW f32 -> NHWC bf16 via 32x32 LDS tile (both sides coalesced)
__global__ __launch_bounds__(256) void xpose_kernel(const float* __restrict__ x,
                                                    bf16* __restrict__ xT) {
  __shared__ float t[32][33];
  int b = blockIdx.z;
  int c0 = blockIdx.y * 32;
  int p0 = blockIdx.x * 32;
  int lx = threadIdx.x & 31, ly = threadIdx.x >> 5;
  const float* xb = x + ((size_t)b * 512) * 4096;
#pragma unroll
  for (int i = 0; i < 4; ++i)
    t[ly + i * 8][lx] = xb[(size_t)(c0 + ly + i * 8) * 4096 + p0 + lx];
  __syncthreads();
  bf16* xtb = xT + ((size_t)b * 4096) * 512;
#pragma unroll
  for (int i = 0; i < 4; ++i)
    xtb[(size_t)(p0 + ly + i * 8) * 512 + c0 + lx] = (bf16)t[lx][ly + i * 8];
}

__global__ __launch_bounds__(256) void f2bf_kernel(const float* __restrict__ in,
                                                   bf16* __restrict__ out, int n) {
  int i = blockIdx.x * 256 + threadIdx.x;
  if (i < n) out[i] = (bf16)in[i];
}

__global__ __launch_bounds__(256) void repack_w2_kernel(const float* __restrict__ w,
                                                        bf16* __restrict__ out) {
  int i = blockIdx.x * 256 + threadIdx.x;   // dest = (t*512+co)*512 + c
  if (i >= 9 * 512 * 512) return;
  int t = i / (512 * 512);
  int rem = i % (512 * 512);
  int co = rem >> 9, c = rem & 511;
  out[i] = (bf16)w[((size_t)co * 512 + c) * 9 + t];
}

__global__ __launch_bounds__(256) void repack_wdw_kernel(const float* __restrict__ w,
                                                         float* __restrict__ out) {
  int i = blockIdx.x * 256 + threadIdx.x;   // dest = tap*512 + c
  if (i >= 64 * 512) return;
  int tap = i >> 9, c = i & 511;
  out[i] = w[c * 64 + tap];
}

// ---------------------------------------------------------------------------
extern "C" void kernel_launch(void* const* d_in, const int* in_sizes, int n_in,
                              void* d_out, int out_size, void* d_ws, size_t ws_size,
                              hipStream_t stream) {
  const float* x    = (const float*)d_in[0];
  const float* w1   = (const float*)d_in[1];
  const float* g1   = (const float*)d_in[2];
  const float* b1   = (const float*)d_in[3];
  const float* w2   = (const float*)d_in[4];
  const float* g2   = (const float*)d_in[5];
  const float* b2   = (const float*)d_in[6];
  const float* wqkv = (const float*)d_in[7];
  const float* rel  = (const float*)d_in[8];
  const float* wdw  = (const float*)d_in[9];
  const float* g3   = (const float*)d_in[10];
  const float* b3   = (const float*)d_in[11];
  const float* wpw  = (const float*)d_in[12];

  char* ws = (char*)d_ws;
  size_t off = 0;
  auto take = [&](size_t bytes) -> char* {
    char* p = ws + off;
    off = (off + bytes + 255) & ~(size_t)255;
    return p;
  };
  const size_t NPX = 16777216ull;                  // 8*512*4096
  bf16* xT     = (bf16*)take(NPX * 2);             // NHWC activations
  bf16* w1bf   = (bf16*)take(262144ull * 2);
  bf16* w2bf   = (bf16*)take(2359296ull * 2);
  bf16* wqbf   = (bf16*)take(786432ull * 2);
  bf16* wpbf   = (bf16*)take(262144ull * 2);
  float* wdwT  = (float*)take(32768ull * 4);
  float* localT = (float*)take(NPX * 4);
  float* attT   = (float*)take(NPX * 4);
  char* big     = take(100663296ull);              // qkv NHWC bf16
  bf16* qkvT    = (bf16*)big;
  float* mixT   = (float*)big;                     // reuse after attention
  bf16* dwT     = (bf16*)(big + 67108864ull);      // reuse tail of qkv region

  // 1) layout conversion / weight repack
  xpose_kernel<<<dim3(128, 16, 8), 256, 0, stream>>>(x, xT);
  f2bf_kernel<<<1024, 256, 0, stream>>>(w1, w1bf, 262144);
  f2bf_kernel<<<3072, 256, 0, stream>>>(wqkv, wqbf, 786432);
  f2bf_kernel<<<1024, 256, 0, stream>>>(wpw, wpbf, 262144);
  repack_w2_kernel<<<9216, 256, 0, stream>>>(w2, w2bf);
  repack_wdw_kernel<<<128, 256, 0, stream>>>(wdw, wdwT);

  // 2) local branch: BN1(1x1) store, then += BN2(3x3 implicit GEMM)
  gemm_kernel<2, 1><<<dim3(64, 4, 8), 256, 0, stream>>>(
      w1bf, xT, localT, nullptr, g1, b1, 512);
  gemm_kernel<3, 9><<<dim3(64, 4, 8), 256, 0, stream>>>(
      w2bf, xT, localT, nullptr, g2, b2, 512);

  // 3) qkv projection -> NHWC bf16
  gemm_kernel<1, 1><<<dim3(64, 12, 8), 256, 0, stream>>>(
      wqbf, xT, nullptr, qkvT, nullptr, nullptr, 1536);

  // 4) window attention (512 windows x 16 heads)
  attn_kernel<<<dim3(512, 16), 128, 0, stream>>>(qkvT, rel, attT);

  // 5) dual 8-wide avg-pool + local  (NHWC, coalesced)
  pool_kernel<<<(int)(NPX / 256), 256, 0, stream>>>(attT, localT, mixT);

  // 6) depthwise 8x8 conv + BN3 -> NHWC bf16
  dwconv_kernel<<<(int)(8ull * 4096 * 128 / 256), 256, 0, stream>>>(
      mixT, wdwT, g3, b3, dwT);

  // 7) final pointwise 1x1 conv -> d_out (NCHW fp32)
  gemm_kernel<0, 1><<<dim3(64, 4, 8), 256, 0, stream>>>(
      wpbf, dwT, (float*)d_out, nullptr, nullptr, nullptr, 512);
}
